// _LSTMLayer_37778532335857
// MI455X (gfx1250) — matrix operand order repaired and checked by reading, not compiled, not run on hardware
//
#include <hip/hip_runtime.h>
#include <hip/hip_bf16.h>
#include <math.h>

typedef __bf16 bf16_t;
typedef __attribute__((ext_vector_type(16))) __bf16 v16bf;
typedef __attribute__((ext_vector_type(8)))  __bf16 v8bf;
typedef __attribute__((ext_vector_type(8)))  float  v8f;

#define B_  64
#define T_  512
#define D_  1024
#define H_  1024
#define G4_ 4096   /* 4*H */

static __device__ __forceinline__ float sigmoidf_(float x) {
    return 1.0f / (1.0f + __expf(-x));
}

// ---------------------------------------------------------------------------
// Prep: bf16 copies of x, transposed bf16 Wi/Wh (N-major so WMMA B-fragments
// are contiguous 32B loads), bf16 h0, and zeroed per-step barrier counters.
// ---------------------------------------------------------------------------
__global__ void lstm_convert_kernel(const float* __restrict__ x,
                                    const float* __restrict__ Wi,
                                    const float* __restrict__ Wh,
                                    const float* __restrict__ h0,
                                    bf16_t* __restrict__ xbf,
                                    bf16_t* __restrict__ WiT,
                                    bf16_t* __restrict__ WhT,
                                    bf16_t* __restrict__ hbf,
                                    unsigned* __restrict__ cnt) {
    size_t i      = (size_t)blockIdx.x * blockDim.x + threadIdx.x;
    size_t stride = (size_t)gridDim.x * blockDim.x;

    const size_t NX = (size_t)B_ * T_ * D_;
    for (size_t j = i; j < NX; j += stride) xbf[j] = (bf16_t)x[j];

    const size_t NW = (size_t)G4_ * D_;            // WiT/WhT: [4H][K]
    for (size_t j = i; j < NW; j += stride) {
        size_t n = j >> 10, k = j & 1023;          // D_ == H_ == 1024
        WiT[j] = (bf16_t)Wi[k * G4_ + n];
        WhT[j] = (bf16_t)Wh[k * G4_ + n];
    }

    const size_t NH = (size_t)B_ * H_;
    for (size_t j = i; j < NH; j += stride) hbf[j] = (bf16_t)h0[j];

    for (size_t j = i; j < (size_t)T_; j += stride) cnt[j] = 0u;
}

// ---------------------------------------------------------------------------
// Phase 1: pregates[t][b][n] = x[b,t,:] @ Wi[:,n] + bias[n]
// M = B*T = 32768, N = 4096, K = 1024. Block = 8 waves; each wave computes a
// 16x64 strip (4 accumulators), bias pre-seeded into C.
// ---------------------------------------------------------------------------
__global__ __launch_bounds__(256)
void lstm_xgemm_kernel(const bf16_t* __restrict__ xbf,
                       const bf16_t* __restrict__ WiT,
                       const float*  __restrict__ bias,
                       float*        __restrict__ pg) {
    const int lane = threadIdx.x & 31;
    const int wave = threadIdx.x >> 5;
    const int n0   = blockIdx.x * 64;                 // N tile base
    const int m0   = blockIdx.y * 128 + wave * 16;    // M tile base
    const int hi8  = lane >> 4;                       // 0 | 1
    const int l15  = lane & 15;

    v8f acc[4];
    for (int nt = 0; nt < 4; ++nt) {
        float bv = bias[n0 + nt * 16 + l15];
        for (int r = 0; r < 8; ++r) acc[nt][r] = bv;
    }

    const bf16_t* arow = xbf + (size_t)(m0 + l15) * D_;
    for (int k0 = 0; k0 < D_; k0 += 32) {
        // A fragment (contiguous, matches B): lanes<16 hold K=k0..k0+15;
        // lanes>=16 hold K=k0+16..k0+31
        v16bf a = *(const v16bf*)(arow + k0 + hi8 * 16);

        int koff = k0 + hi8 * 16;                     // B: K=k0..15 / k0+16..31
        for (int nt = 0; nt < 4; ++nt) {
            const bf16_t* bp = WiT + (size_t)(n0 + nt * 16 + l15) * D_ + koff;
            v16bf bfrag = *(const v16bf*)bp;
            acc[nt] = __builtin_amdgcn_wmma_f32_16x16x32_bf16(
                false, a, false, bfrag, (short)0, acc[nt], false, false);
        }
    }

    // C/D layout: VGPR r -> row m0+r (lanes 0-15) / m0+r+8 (lanes 16-31)
    for (int nt = 0; nt < 4; ++nt) {
        int n = n0 + nt * 16 + l15;
        for (int r = 0; r < 8; ++r) {
            int gr = m0 + r + hi8 * 8;                // row in [B*T): b*T + t
            int t  = gr & (T_ - 1);
            int b  = gr >> 9;
            pg[((size_t)t * B_ + b) * G4_ + n] = acc[nt][r];
        }
    }
}

// ---------------------------------------------------------------------------
// Phase 2: persistent cooperative recurrence. 64 blocks; block bx owns h
// columns [bx*16, bx*16+16) for the whole sequence (c stays in registers).
// Per step: gates(64x64) = pregate + h @ Wh via WMMA (wave w: m-tile w&3,
// gate pair (w>>2)*2), LDS cross-wave gate combine, elementwise cell update,
// write bf16 h + f32 ys, then one device-scope barrier.
// ---------------------------------------------------------------------------
__global__ __launch_bounds__(256)
void lstm_recurrent_kernel(const float*  __restrict__ c0,
                           const float*  __restrict__ pg,
                           const bf16_t* __restrict__ WhT,
                           bf16_t*       __restrict__ hbf,
                           unsigned*     __restrict__ cnt,
                           float*        __restrict__ out,
                           int nblocks) {
    __shared__ float lds_g[4 * 64 * 16];              // 16 KB: [gate][b][hc]

    const int tid  = threadIdx.x;
    const int lane = tid & 31;
    const int wave = tid >> 5;
    const int hc0  = blockIdx.x * 16;                 // owned h-column base
    const int hi8  = lane >> 4;
    const int l15  = lane & 15;
    const int mt   = wave & 3;                        // batch-row tile
    const int gp   = (wave >> 2) * 2;                 // gate pair base

    // persistent cell state: 4 (b, hc) elements per thread
    float creg[4];
    for (int j = 0; j < 4; ++j) {
        int e = tid + 256 * j;
        int b = e >> 4, hc = e & 15;
        creg[j] = c0[(size_t)b * H_ + hc0 + hc];
    }

    float* ys = out + 2 * (size_t)B_ * H_;

    const bf16_t* bptr0 = WhT + (size_t)((gp    ) * H_ + hc0 + l15) * H_;
    const bf16_t* bptr1 = WhT + (size_t)((gp + 1) * H_ + hc0 + l15) * H_;
    const bf16_t* arow  = hbf + (size_t)(mt * 16 + l15) * H_;

    for (int t = 0; t < T_; ++t) {
        // ---- accumulators seeded with pregates (x@Wi + bias) ----
        const float* pgt = pg + (size_t)t * B_ * G4_;
        v8f acc0, acc1;
        for (int r = 0; r < 8; ++r) {
            int b = mt * 16 + r + hi8 * 8;
            acc0[r] = pgt[(size_t)b * G4_ + (gp    ) * H_ + hc0 + l15];
            acc1[r] = pgt[(size_t)b * G4_ + (gp + 1) * H_ + hc0 + l15];
        }

        // ---- h @ Wh : K=1024 in chunks of 32 (bf16 WMMA, f32 accum) ----
        for (int k0 = 0; k0 < H_; k0 += 32) {
            v16bf a = *(const v16bf*)(arow + k0 + hi8 * 16);
            int koff = k0 + hi8 * 16;
            v16bf b0 = *(const v16bf*)(bptr0 + koff);
            v16bf b1 = *(const v16bf*)(bptr1 + koff);
            acc0 = __builtin_amdgcn_wmma_f32_16x16x32_bf16(
                false, a, false, b0, (short)0, acc0, false, false);
            acc1 = __builtin_amdgcn_wmma_f32_16x16x32_bf16(
                false, a, false, b1, (short)0, acc1, false, false);
        }

        // ---- cross-wave gate combine through LDS ----
        for (int r = 0; r < 8; ++r) {
            int b = mt * 16 + r + hi8 * 8;
            lds_g[((gp    ) * 64 + b) * 16 + l15] = acc0[r];
            lds_g[((gp + 1) * 64 + b) * 16 + l15] = acc1[r];
        }
        __syncthreads();

        // ---- elementwise LSTM cell update (c kept in registers) ----
        for (int j = 0; j < 4; ++j) {
            int e = tid + 256 * j;
            int b = e >> 4, hc = e & 15;
            float iv = lds_g[(0 * 64 + b) * 16 + hc];
            float fv = lds_g[(1 * 64 + b) * 16 + hc];
            float gv = lds_g[(2 * 64 + b) * 16 + hc];
            float ov = lds_g[(3 * 64 + b) * 16 + hc];
            float nc = sigmoidf_(fv) * creg[j] + sigmoidf_(iv) * tanhf(gv);
            float nh = sigmoidf_(ov) * tanhf(nc);
            creg[j] = nc;
            hbf[(size_t)b * H_ + hc0 + hc]              = (bf16_t)nh;
            ys[((size_t)b * T_ + t) * H_ + hc0 + hc]    = nh;
            if (t == T_ - 1) {
                out[(size_t)b * H_ + hc0 + hc]                        = nc;
                out[(size_t)B_ * H_ + (size_t)b * H_ + hc0 + hc]      = nh;
            }
        }

        // ---- device-scope barrier: new h visible to every block ----
        __threadfence();
        __syncthreads();
        if (tid == 0) {
            __hip_atomic_fetch_add(&cnt[t], 1u, __ATOMIC_ACQ_REL,
                                   __HIP_MEMORY_SCOPE_AGENT);
            while (__hip_atomic_load(&cnt[t], __ATOMIC_ACQUIRE,
                                     __HIP_MEMORY_SCOPE_AGENT) < (unsigned)nblocks) {
                __builtin_amdgcn_s_sleep(1);
            }
        }
        __syncthreads();
        __builtin_amdgcn_fence(__ATOMIC_ACQUIRE, "agent");
    }
}

// ---------------------------------------------------------------------------
extern "C" void kernel_launch(void* const* d_in, const int* in_sizes, int n_in,
                              void* d_out, int out_size, void* d_ws, size_t ws_size,
                              hipStream_t stream) {
    (void)in_sizes; (void)n_in; (void)out_size; (void)ws_size;

    const float* c0   = (const float*)d_in[0];
    const float* h0   = (const float*)d_in[1];
    const float* x    = (const float*)d_in[2];
    const float* Wi   = (const float*)d_in[3];
    const float* Wh   = (const float*)d_in[4];
    const float* bias = (const float*)d_in[5];

    char* ws = (char*)d_ws;
    size_t off = 0;
    bf16_t* xbf = (bf16_t*)(ws + off); off += (size_t)2 * B_ * T_ * D_;   // 64 MiB
    bf16_t* WiT = (bf16_t*)(ws + off); off += (size_t)2 * G4_ * D_;       //  8 MiB
    bf16_t* WhT = (bf16_t*)(ws + off); off += (size_t)2 * G4_ * H_;       //  8 MiB
    float*  pg  = (float* )(ws + off); off += (size_t)4 * T_ * B_ * G4_;  // 512 MiB
    bf16_t* hbf = (bf16_t*)(ws + off); off += (size_t)2 * B_ * H_;        // 128 KiB
    unsigned* cnt = (unsigned*)(ws + off); off += (size_t)4 * T_;         //   2 KiB

    lstm_convert_kernel<<<4096, 256, 0, stream>>>(x, Wi, Wh, h0,
                                                  xbf, WiT, WhT, hbf, cnt);

    dim3 g1(G4_ / 64, (B_ * T_) / 128);   // (64, 256)
    lstm_xgemm_kernel<<<g1, 256, 0, stream>>>(xbf, WiT, bias, pg);

    lstm_recurrent_kernel<<<H_ / 16, 256, 0, stream>>>(c0, pg, WhT, hbf, cnt,
                                                       (float*)d_out, H_ / 16);
}
